// RNN_87900800680688
// MI455X (gfx1250) — matrix operand order repaired and checked by reading, not compiled, not run on hardware
//
#include <hip/hip_runtime.h>
#include <math.h>

// ---------------- types for CDNA5 WMMA (gfx1250, wave32) ----------------
typedef __bf16 bf16_t;
typedef bf16_t v16bf __attribute__((ext_vector_type(16)));
typedef bf16_t v8bf  __attribute__((ext_vector_type(8)));
typedef bf16_t v4bf  __attribute__((ext_vector_type(4)));
typedef float  v8f   __attribute__((ext_vector_type(8)));

union Frag {
    v16bf v;
    v8bf  h[2];
};

// Problem sizes (fixed by the reference)
#define SEQ     512
#define BATCH   256
#define DIM_IN  512
#define DIM_LAT 1024

#define AS_STRIDE 520    // 512 + 8 pad  -> 1040B row stride = 4 banks mod 64
#define BS_STRIDE 1032   // 1024 + 8 pad -> 2064B row stride = 4 banks mod 64

// persistent-step geometry: 64 blocks = 8 (M/32) x 8 (N/128)
#define STEP_BLOCKS 64

// =====================================================================
// ws layout (bf16 elements):
//   [0,               524288)   W_i  (1024x512)
//   [524288,         1572864)   W_h  (1024x1024)
//   [1572864,        1835008)   H ping  (256x1024)
//   [1835008,        2097152)   H pong  (256x1024)
//   byte offset 4 MiB:          u32 global-barrier counter
// =====================================================================
#define WS_WI   0
#define WS_WH   (1024*512)
#define WS_H0   (WS_WH + 1024*1024)
#define WS_H1   (WS_H0 + 256*1024)
#define WS_TOT  (WS_H1 + 256*1024)

// ---- CDNA5 async copy: 16B global -> LDS, tracked by ASYNCcnt ----
__device__ __forceinline__ void async_copy_b128(unsigned lds_byte_off,
                                                const void* sbase,
                                                unsigned voff_bytes) {
    asm volatile("global_load_async_to_lds_b128 %0, %1, %2"
                 :: "v"(lds_byte_off), "v"(voff_bytes), "s"(sbase)
                 : "memory");
}
__device__ __forceinline__ void wait_asynccnt0() {
    asm volatile("s_wait_asynccnt 0x0" ::: "memory");
}
__device__ __forceinline__ unsigned lds_off(const void* p) {
    // generic LDS pointer: low 32 bits are the LDS byte address (aperture map)
    return (unsigned)(uintptr_t)p;
}

// =====================================================================
// prep: W_i, W_h -> bf16 in ws; zero bf16 h0 ping buffer; zero barrier ctr
// =====================================================================
__global__ __launch_bounds__(256) void prep_kernel(const float* __restrict__ Wi,
                                                   const float* __restrict__ Wh,
                                                   bf16_t* __restrict__ ws,
                                                   unsigned* __restrict__ ctr) {
    size_t i = (size_t)blockIdx.x * 256 + threadIdx.x;
    if (i == 0) *ctr = 0u;
    const size_t nwi = 1024 * 512;
    const size_t nwh = 1024 * 1024;
    const size_t nh  = 256 * 1024;
    if (i < nwi) {
        ws[WS_WI + i] = (bf16_t)Wi[i];
    } else if (i < nwi + nwh) {
        ws[WS_WH + (i - nwi)] = (bf16_t)Wh[i - nwi];
    } else if (i < nwi + nwh + nh) {
        ws[WS_H0 + (i - nwi - nwh)] = (bf16_t)0.0f;
    }
}

// =====================================================================
// proj: xi[m,n] = sum_k x[m,k]*Wi[n,k]   M=SEQ*BATCH, K=512, N=1024
// Block: 32 M-rows x all N. x staged once (full K) in LDS -> x read from
// HBM exactly once. 8 waves = 2(M) x 4(N); wave = 16 rows x 256 cols as
// 4 chunks of 4 16x16 tiles. HBM-bound: ~0.8 GB total -> ~35us roofline.
// =====================================================================
__global__ __launch_bounds__(256) void proj_kernel(const float* __restrict__ x,
                                                   const bf16_t* __restrict__ Wi,
                                                   float* __restrict__ xi) {
    __shared__ __align__(16) bf16_t As[32 * AS_STRIDE];

    const int tid = threadIdx.x;
    const size_t r0 = (size_t)blockIdx.x * 32;

    const float4* xg = (const float4*)x; // 128 float4 per row
    for (int q = tid; q < 32 * 128; q += 256) {
        const int row = q >> 7;
        const int qq  = q & 127;
        float4 f = xg[(r0 + row) * 128 + qq];
        v4bf b4;
        b4.x = (bf16_t)f.x; b4.y = (bf16_t)f.y;
        b4.z = (bf16_t)f.z; b4.w = (bf16_t)f.w;
        *(v4bf*)&As[row * AS_STRIDE + qq * 4] = b4;
    }
    __syncthreads();

    const int lane = tid & 31;
    const int ll   = lane & 15;
    const int g    = lane >> 4;
    const int w    = tid >> 5;
    const int mw   = w >> 2;          // 0..1
    const int nw   = w & 3;           // 0..3

    const bf16_t* arow = &As[(mw * 16 + ll) * AS_STRIDE];

    for (int nc = 0; nc < 4; ++nc) {
        v8f acc[4] = {};
        for (int ks = 0; ks < 16; ++ks) {
            const int k0 = ks * 32;
            Frag a;
            a.h[0] = *(const v8bf*)(arow + k0 + 8 * g);
            a.h[1] = *(const v8bf*)(arow + k0 + 16 + 8 * g);
#pragma unroll
            for (int j = 0; j < 4; ++j) {
                const int n = nw * 256 + nc * 64 + j * 16 + ll;
                const bf16_t* brow = &Wi[(size_t)n * DIM_IN + k0 + 8 * g];
                Frag b;
                b.h[0] = *(const v8bf*)(brow);
                b.h[1] = *(const v8bf*)(brow + 16);
                acc[j] = __builtin_amdgcn_wmma_f32_16x16x32_bf16(
                    false, a.v, false, b.v, (short)0, acc[j], false, false);
            }
        }
#pragma unroll
        for (int j = 0; j < 4; ++j) {
            const int n0 = nw * 256 + nc * 64 + j * 16;
#pragma unroll
            for (int vv = 0; vv < 8; ++vv) {
                const size_t row = r0 + mw * 16 + vv + 8 * g;
                xi[row * DIM_LAT + n0 + ll] = acc[j][vv];
            }
        }
    }
}

// =====================================================================
// persistent recurrence: all 512 steps in one kernel.
//   Grid: 64 blocks = 8 (batch/32) x 8 (latent/128), 256 thr (8 waves).
//   Per block: pin its W_h slice (128x1024 bf16, ~258KB padded) in LDS
//   ONCE via async-to-LDS; per step async-stage the 32x1024 bf16 h_{t-1}
//   slice (two K halves). Device-scope atomic barrier between steps.
//   Epilogue: h = tanh(acc + xi + b) -> hs[t] f32 in place, bf16 ping-pong,
//   and h_final on the last step.
// =====================================================================
__global__ __launch_bounds__(256) void rnn_persistent(
        const bf16_t* __restrict__ Wh,
        const float*  __restrict__ bh,
        float* __restrict__ hs,
        float* __restrict__ hfinal,
        bf16_t* __restrict__ H0,
        bf16_t* __restrict__ H1,
        unsigned* __restrict__ ctr) {
    extern __shared__ __align__(16) char smem[];
    bf16_t* Bs = (bf16_t*)smem;                                   // [128][BS_STRIDE]
    bf16_t* As = (bf16_t*)(smem + (size_t)128 * BS_STRIDE * 2);   // [32][AS_STRIDE]

    const int tid = threadIdx.x;
    const int mg  = blockIdx.x & 7;          // batch-row group
    const int ngi = blockIdx.x >> 3;         // latent-col group
    const int r0  = mg * 32;
    const int ng  = ngi * 128;

    const int lane = tid & 31;
    const int ll   = lane & 15;
    const int g    = lane >> 4;
    const int w    = tid >> 5;
    const int mw   = w >> 2;                 // 0..1 (16-row group)
    const int nw   = w & 3;                  // 0..3 (32-col group)

    // ---- one-time: pin W_h rows [ng, ng+128) x K=1024 in LDS (async) ----
    for (int q = tid; q < 128 * 128; q += 256) {      // 16B chunks
        const int row = q >> 7;
        const int ko  = (q & 127) << 3;               // bf16 elements
        async_copy_b128(lds_off(&Bs[row * BS_STRIDE + ko]), Wh,
                        (unsigned)(((ng + row) * DIM_LAT + ko) * sizeof(bf16_t)));
    }
    wait_asynccnt0();
    __syncthreads();

    for (int t = 0; t < SEQ; ++t) {
        const bf16_t* Hp = (t & 1) ? H1 : H0;
        bf16_t*       Hc = (t & 1) ? H0 : H1;
        float*        hst = hs + (size_t)t * BATCH * DIM_LAT;

        v8f acc[2] = {};
        for (int kb = 0; kb < 2; ++kb) {
            const int kglob = kb * 512;
            __syncthreads();
            // async-stage 32 x 512 bf16 slice of h_{t-1}
            for (int q = tid; q < 32 * 64; q += 256) {
                const int row = q >> 6;
                const int ko  = (q & 63) << 3;
                async_copy_b128(lds_off(&As[row * AS_STRIDE + ko]), Hp,
                                (unsigned)(((r0 + row) * DIM_LAT + kglob + ko) * sizeof(bf16_t)));
            }
            wait_asynccnt0();
            __syncthreads();

            const bf16_t* arow = &As[(mw * 16 + ll) * AS_STRIDE];
            for (int ks = 0; ks < 16; ++ks) {
                const int k0 = ks * 32;
                Frag a;
                a.h[0] = *(const v8bf*)(arow + k0 + 8 * g);
                a.h[1] = *(const v8bf*)(arow + k0 + 16 + 8 * g);
#pragma unroll
                for (int j = 0; j < 2; ++j) {
                    const bf16_t* brow =
                        &Bs[(size_t)(nw * 32 + j * 16 + ll) * BS_STRIDE + kglob + k0 + 8 * g];
                    Frag b;
                    b.h[0] = *(const v8bf*)(brow);
                    b.h[1] = *(const v8bf*)(brow + 16);
                    acc[j] = __builtin_amdgcn_wmma_f32_16x16x32_bf16(
                        false, a.v, false, b.v, (short)0, acc[j], false, false);
                }
            }
        }

        // epilogue
#pragma unroll
        for (int j = 0; j < 2; ++j) {
            const int col = ng + nw * 32 + j * 16 + ll;
            const float bias = bh[col];
#pragma unroll
            for (int vv = 0; vv < 8; ++vv) {
                const int brow = r0 + mw * 16 + vv + 8 * g;
                const size_t idx = (size_t)brow * DIM_LAT + col;
                const float hv = tanhf(acc[j][vv] + hst[idx] + bias);
                hst[idx] = hv;
                Hc[idx]  = (bf16_t)hv;
                if (t == SEQ - 1) hfinal[idx] = hv;
            }
        }

        // ---- device-scope barrier between steps ----
        if (t < SEQ - 1) {
            __syncthreads();
            if (tid == 0) {
                __hip_atomic_fetch_add(ctr, 1u, __ATOMIC_RELEASE,
                                       __HIP_MEMORY_SCOPE_AGENT);
                const unsigned target = (unsigned)STEP_BLOCKS * (unsigned)(t + 1);
                while (__hip_atomic_load(ctr, __ATOMIC_ACQUIRE,
                                         __HIP_MEMORY_SCOPE_AGENT) < target) {
                    __builtin_amdgcn_s_sleep(2);
                }
            }
            __syncthreads();
        }
    }
}

// =====================================================================
// launcher
// inputs: x [512,256,512] f32 | W_i [1024,512] | W_h [1024,1024] | b_h [1024]
// d_out: h_final (256*1024 f32) then hs (512*256*1024 f32)
// =====================================================================
extern "C" void kernel_launch(void* const* d_in, const int* in_sizes, int n_in,
                              void* d_out, int out_size, void* d_ws, size_t ws_size,
                              hipStream_t stream) {
    const float* x  = (const float*)d_in[0];
    const float* Wi = (const float*)d_in[1];
    const float* Wh = (const float*)d_in[2];
    const float* bh = (const float*)d_in[3];

    float* out    = (float*)d_out;
    float* hfinal = out;
    float* hs     = out + (size_t)BATCH * DIM_LAT;

    bf16_t*   ws   = (bf16_t*)d_ws;                       // needs 4 MiB + 4
    bf16_t*   Wi_b = ws + WS_WI;
    bf16_t*   Wh_b = ws + WS_WH;
    bf16_t*   H0   = ws + WS_H0;
    bf16_t*   H1   = ws + WS_H1;
    unsigned* ctr  = (unsigned*)((char*)d_ws + (size_t)WS_TOT * sizeof(bf16_t));

    // 1) weights -> bf16, zero h0 + barrier counter
    {
        const size_t total = (size_t)1024 * 512 + (size_t)1024 * 1024 + (size_t)256 * 1024;
        const int grid = (int)((total + 255) / 256);
        prep_kernel<<<grid, 256, 0, stream>>>(Wi, Wh, ws, ctr);
    }

    // 2) xi = x @ Wi^T (written into hs region, f32)
    proj_kernel<<<(SEQ * BATCH) / 32, 256, 0, stream>>>(x, Wi_b, hs);

    // 3) all 512 recurrent steps in one persistent kernel
    {
        const size_t lds_bytes =
            ((size_t)128 * BS_STRIDE + (size_t)32 * AS_STRIDE) * sizeof(bf16_t); // ~291 KB
        rnn_persistent<<<STEP_BLOCKS, 256, lds_bytes, stream>>>(
            Wh_b, bh, hs, hfinal, H0, H1, ctr);
    }
}
